// model_attention_final_37443524886530
// MI455X (gfx1250) — matrix operand order repaired and checked by reading, not compiled, not run on hardware
//
#include <hip/hip_runtime.h>
#include <hip/hip_bf16.h>

// MI455X / gfx1250, wave32, WMMA f32_16x16x32_f16.
// One workgroup (8 wave32) per (n,h); 16-channel chain fused in LDS.
// All WMMA fragments load as 2x ds_load_b128 (transposes are materialized at
// store time, which is free since those stores are element-wise anyway).

typedef __attribute__((ext_vector_type(16))) _Float16     v16h;
typedef __attribute__((ext_vector_type(8)))  float        v8f;
typedef __attribute__((ext_vector_type(4)))  unsigned int v4u;

#define PIT 72   // f16 (and f32) element pitch: 144B rows, 16B-aligned, conflict-free

// LDS layout (bytes); every region 16B aligned.
constexpr int MATH  = 64 * PIT * 2;          // 9216  (f16 64x64, pitch 72)
constexpr int MATF  = 64 * PIT * 4;          // 18432 (f32)
constexpr int OFF_COR  = 0;                  // 16ch * 64*64 f16 (pitch 64) = 131072
constexpr int OFF_Q    = 131072;             // Q   row-major
constexpr int OFF_KT   = OFF_Q    + MATH;    // K^T row-major
constexpr int OFF_VT   = OFF_KT   + MATH;    // V^T row-major
constexpr int OFF_QKT  = OFF_VT   + MATH;    // (Q@K)^T row-major
constexpr int OFF_A    = OFF_QKT  + MATH;    // A   row-major
constexpr int OFF_M1   = OFF_A    + MATH;    // A@V row-major
constexpr int OFF_APRE = OFF_M1   + MATH;    // f32
constexpr int OFF_INV  = OFF_APRE + MATF;    // 16*64 f32 = 4096
constexpr int OFF_W    = OFF_INV  + 4096;    // 3*256 f32 = 3072
constexpr int OFF_RED  = OFF_W    + 3072;    // 256 f32
constexpr int OFF_DIA  = OFF_RED  + 1024;    // 64 f32
constexpr int OFF_INVD = OFF_DIA  + 256;     // 64 f32 (+pad)
constexpr int SMEM_BYTES = OFF_INVD + 256;   // 213504 <= 320KB/WGP

union FragH { v16h h; v4u q[2]; };
union PackH { _Float16 h[8]; v4u q; };
union Pack2 { _Float16 h[2]; unsigned int u; };

// A-operand 16x32 f16 fragment from row-major src (ISA 7.12.2 layout):
// lane lg holds row m0+lg; VGPR v<4: K=2v+8*half; v>=4: K=16+2(v-4)+8*half.
// dwords: [36*row + k0/2 + 4*half + 0..3] and [... + 8..11] -> 2x b128.
__device__ __forceinline__ v16h load_a_fast(const _Float16* src, int m0, int k0,
                                            int lg, int half) {
  FragH f;
  const v4u* qp = (const v4u*)src;
  const int base = 9 * (m0 + lg) + (k0 >> 3) + half;   // uint4 index
  f.q[0] = qp[base];
  f.q[1] = qp[base + 2];
  return f.h;
}

// B-operand 32x16 f16 fragment, where srcT holds B^T row-major (srcT[n][k]=B[k][n]):
// lane lg holds col n0+lg; VGPR v<4: K=k0+2v+8*half; v>=4: K=k0+16+2(v-4)+8*half.
__device__ __forceinline__ v16h load_b_fast(const _Float16* srcT, int k0, int n0,
                                            int lg, int half) {
  FragH f;
  const v4u* qp = (const v4u*)srcT;
  const int base = 9 * (n0 + lg) + (k0 >> 3) + half;
  f.q[0] = qp[base];
  f.q[1] = qp[base + 2];
  return f.h;
}

// 16x16 tile of a 64x64 matmul: 2 x v_wmma_f32_16x16x32_f16 over K=64.
__device__ __forceinline__ v8f mm64_tile(const _Float16* A, const _Float16* Bt,
                                         int m0, int n0, int lg, int half) {
  v8f acc = {};
#pragma unroll
  for (int k0 = 0; k0 < 64; k0 += 32) {
    v16h a = load_a_fast(A, m0, k0, lg, half);
    v16h b = load_b_fast(Bt, k0, n0, lg, half);
    acc = __builtin_amdgcn_wmma_f32_16x16x32_f16(false, a, false, b,
                                                 (short)0, acc, false, false);
  }
  return acc;
}

// D layout: VGPR r -> row m0+r+8*half, col n0+lg.
// Transposed store: lane's 8 rows become 8 contiguous f16 in dst^T -> one b128.
__device__ __forceinline__ void store_tile_tr(_Float16* dstT, int m0, int n0,
                                              v8f acc, int lg, int half) {
  PackH pk;
#pragma unroll
  for (int r2 = 0; r2 < 8; ++r2) pk.h[r2] = (_Float16)acc[r2];
  v4u* qp = (v4u*)dstT;
  qp[9 * (n0 + lg) + (m0 >> 3) + half] = pk.q;
}
__device__ __forceinline__ void store_tile_h(_Float16* dst, int m0, int n0,
                                             v8f acc, int lg, int half) {
#pragma unroll
  for (int r2 = 0; r2 < 8; ++r2)
    dst[(m0 + r2 + 8 * half) * PIT + n0 + lg] = (_Float16)acc[r2];
}
__device__ __forceinline__ void store_tile_f(float* dst, int m0, int n0,
                                             v8f acc, int lg, int half) {
#pragma unroll
  for (int r2 = 0; r2 < 8; ++r2)
    dst[(m0 + r2 + 8 * half) * PIT + n0 + lg] = acc[r2];
}

__global__ __launch_bounds__(256)
void spd_attn_kernel(const float* __restrict__ x,
                     const float* __restrict__ wq,
                     const float* __restrict__ wk,
                     const float* __restrict__ wv,
                     float* __restrict__ out) {
  extern __shared__ char smem[];
  _Float16* cor  = (_Float16*)(smem + OFF_COR);   // pitch 64, 16 channels
  _Float16* Qc   = (_Float16*)(smem + OFF_Q);
  _Float16* KcT  = (_Float16*)(smem + OFF_KT);
  _Float16* VcT  = (_Float16*)(smem + OFF_VT);
  _Float16* QKT  = (_Float16*)(smem + OFF_QKT);
  _Float16* As   = (_Float16*)(smem + OFF_A);
  _Float16* M1s  = (_Float16*)(smem + OFF_M1);
  float*    apre = (float*)(smem + OFF_APRE);
  float*    invl = (float*)(smem + OFF_INV);      // [cc*64 + r]
  float*    wql  = (float*)(smem + OFF_W);
  float*    wkl  = wql + 256;
  float*    wvl  = wql + 512;
  float*    red  = (float*)(smem + OFF_RED);
  float*    dia  = (float*)(smem + OFF_DIA);
  float*    invd = (float*)(smem + OFF_INVD);

  const int tid  = threadIdx.x;
  const int wave = tid >> 5;
  const int lane = tid & 31;
  const int lg   = lane & 15;
  const int half = (lane >> 4) & 1;
  const int n    = blockIdx.x >> 2;
  const int h    = blockIdx.x & 3;
  const long inBase = (long)(n * 64 + h * 16) * 4096;  // (n,h,cc=0,0,0)

  // ---- Phase 1: head weights + per-channel diag rsqrt ----
  wql[tid] = wq[h * 256 + tid];
  wkl[tid] = wk[h * 256 + tid];
  wvl[tid] = wv[h * 256 + tid];
#pragma unroll
  for (int i = 0; i < 4; ++i) {
    const int idx = tid + 256 * i;                 // cc*64 + r
    const int cc = idx >> 6, r = idx & 63;
    const float d = fabsf(x[inBase + (long)cc * 4096 + r * 64 + r]);
    invl[idx] = rsqrtf(fmaxf(d, 1e-4f));
  }
  __syncthreads();

  // ---- Phase 2: cor = clip(x * inv_i * inv_j, +-1) -> f16 LDS ----
  for (int i = 0; i < 256; ++i) {
    const int idx = tid + 256 * i;                 // cc*4096 + pix
    const int cc = idx >> 12, pix = idx & 4095;
    const int r = pix >> 6, cl = pix & 63;
    __builtin_prefetch(&x[inBase + idx + 8192], 0, 1);
    float v = x[inBase + idx] * invl[cc * 64 + r] * invl[cc * 64 + cl];
    v = fminf(fmaxf(v, -1.0f), 1.0f);
    cor[idx] = (_Float16)v;
  }
  __syncthreads();

  // ---- Per output channel chain ----
  for (int ch = 0; ch < 16; ++ch) {
    // Mixing over pixel pairs: Q (row-major), K^T, V^T.
    for (int i = 0; i < 8; ++i) {
      const int p2  = (tid + 256 * i) * 2;         // even pixel
      const int row = p2 >> 6, col = p2 & 63;
      float q0 = 0.f, q1 = 0.f, ka = 0.f, kb = 0.f, va = 0.f, vb = 0.f;
#pragma unroll
      for (int cc = 0; cc < 16; ++cc) {
        Pack2 cv; cv.u = *(const unsigned int*)&cor[cc * 4096 + p2];
        const float c0 = (float)cv.h[0], c1 = (float)cv.h[1];
        const float wqv = wql[cc * 16 + ch];
        const float wkv = wkl[cc * 16 + ch];
        const float wvv = wvl[cc * 16 + ch];
        q0 += c0 * wqv; q1 += c1 * wqv;
        ka += c0 * wkv; kb += c1 * wkv;
        va += c0 * wvv; vb += c1 * wvv;
      }
      Pack2 qq; qq.h[0] = (_Float16)q0; qq.h[1] = (_Float16)q1;
      *(unsigned int*)&Qc[row * PIT + col] = qq.u;
      KcT[col * PIT + row]       = (_Float16)ka;
      KcT[(col + 1) * PIT + row] = (_Float16)kb;
      VcT[col * PIT + row]       = (_Float16)va;
      VcT[(col + 1) * PIT + row] = (_Float16)vb;
    }
    __syncthreads();

    // QK = Q @ K  (store transposed, packed b128)
#pragma unroll
    for (int ti = 0; ti < 2; ++ti) {
      const int t = wave + 8 * ti;
      const int m0 = (t >> 2) * 16, n0 = (t & 3) * 16;
      v8f acc = mm64_tile(Qc, KcT, m0, n0, lg, half);
      store_tile_tr(QKT, m0, n0, acc, lg, half);
    }
    __syncthreads();

    // A_pre = K^T @ QK  (f32 for the exp path)
#pragma unroll
    for (int ti = 0; ti < 2; ++ti) {
      const int t = wave + 8 * ti;
      const int m0 = (t >> 2) * 16, n0 = (t & 3) * 16;
      v8f acc = mm64_tile(KcT, QKT, m0, n0, lg, half);
      store_tile_f(apre, m0, n0, acc, lg, half);
    }
    __syncthreads();

    // ---- SPD softmax ----
    float lm = -3.4e38f;
#pragma unroll
    for (int i = 0; i < 16; ++i) {
      const int e = tid * 16 + i;
      lm = fmaxf(lm, apre[(e >> 6) * PIT + (e & 63)]);
    }
    red[tid] = lm;
    __syncthreads();
    for (int s = 128; s > 0; s >>= 1) {
      if (tid < s) red[tid] = fmaxf(red[tid], red[tid + s]);
      __syncthreads();
    }
    const float gmax = red[0];
    __syncthreads();

    {  // exp + row partial sums (4 threads per row)
      const int rr = tid >> 2, c0 = (tid & 3) * 16;
      float ps = 0.f;
#pragma unroll
      for (int i = 0; i < 16; ++i) {
        const float e = expf(apre[rr * PIT + c0 + i] - gmax);
        apre[rr * PIT + c0 + i] = e;
        ps += e;
      }
      red[tid] = ps;
    }
    __syncthreads();
    if (tid < 64)
      dia[tid] = red[4 * tid] + red[4 * tid + 1] + red[4 * tid + 2] + red[4 * tid + 3];
    __syncthreads();
    if (tid < 32) red[tid] = dia[tid] + dia[tid + 32];
    __syncthreads();
    for (int s = 16; s > 0; s >>= 1) {
      if (tid < s) red[tid] += red[tid + s];
      __syncthreads();
    }
    const float S = red[0];
    __syncthreads();
    if (tid < 64)
      invd[tid] = rsqrtf(fmaxf(fmaxf(dia[tid], S * 1e-5f), 1e-5f));
    __syncthreads();
#pragma unroll
    for (int i = 0; i < 16; ++i) {
      const int e = tid * 16 + i;
      const int rr = e >> 6, cl = e & 63;
      As[rr * PIT + cl] = (_Float16)(apre[rr * PIT + cl] * invd[rr] * invd[cl]);
    }
    __syncthreads();

    // M1 = A @ V
#pragma unroll
    for (int ti = 0; ti < 2; ++ti) {
      const int t = wave + 8 * ti;
      const int m0 = (t >> 2) * 16, n0 = (t & 3) * 16;
      v8f acc = mm64_tile(As, VcT, m0, n0, lg, half);
      store_tile_h(M1s, m0, n0, acc, lg, half);
    }
    __syncthreads();

    // out = x + M1 @ A^T  (B-operand reads As row-contiguously)
    const long chBase = inBase + (long)ch * 4096;
#pragma unroll
    for (int ti = 0; ti < 2; ++ti) {
      const int t = wave + 8 * ti;
      const int m0 = (t >> 2) * 16, n0 = (t & 3) * 16;
      v8f acc = mm64_tile(M1s, As, m0, n0, lg, half);
#pragma unroll
      for (int r2 = 0; r2 < 8; ++r2) {
        const int row = m0 + r2 + 8 * half, col = n0 + lg;
        const long gi = chBase + row * 64 + col;
        out[gi] = x[gi] + acc[r2];
      }
    }
    __syncthreads();
  }
}

extern "C" void kernel_launch(void* const* d_in, const int* in_sizes, int n_in,
                              void* d_out, int out_size, void* d_ws, size_t ws_size,
                              hipStream_t stream) {
  const float* x  = (const float*)d_in[0];
  const float* wq = (const float*)d_in[1];
  const float* wk = (const float*)d_in[2];
  const float* wv = (const float*)d_in[3];
  float* out = (float*)d_out;
  (void)in_sizes; (void)n_in; (void)out_size; (void)d_ws; (void)ws_size;

  (void)hipFuncSetAttribute((const void*)spd_attn_kernel,
                            hipFuncAttributeMaxDynamicSharedMemorySize, SMEM_BYTES);
  dim3 grid(512);   // N*H = 128*4
  dim3 block(256);  // 8 wave32
  spd_attn_kernel<<<grid, block, SMEM_BYTES, stream>>>(x, wq, wk, wv, out);
}
